// Attention_35433480192757
// MI455X (gfx1250) — matrix operand order repaired and checked by reading, not compiled, not run on hardware
//
#include <hip/hip_runtime.h>
#include <hip/hip_bf16.h>

typedef __bf16 bf16;
typedef __attribute__((ext_vector_type(8)))  bf16  bf16x8;
typedef __attribute__((ext_vector_type(16))) bf16  bf16x16;
typedef __attribute__((ext_vector_type(8)))  float f32x8;

#define WMMA_BF16(A,B,C) \
  __builtin_amdgcn_wmma_f32_16x16x32_bf16(false,(A),false,(B),(short)0,(C),false,false)

// Async DMA: global (16B per lane) -> LDS, tracked by ASYNCcnt.
__device__ __forceinline__ void async_ld16(unsigned lds_addr, const bf16* gaddr) {
  asm volatile("global_load_async_to_lds_b128 %0, %1, off"
               :: "v"(lds_addr), "v"(gaddr) : "memory");
}
__device__ __forceinline__ void wait_async_le16() {
  asm volatile("s_wait_asynccnt 16" ::: "memory");
}
__device__ __forceinline__ void wait_async_0() {
  asm volatile("s_wait_asynccnt 0" ::: "memory");
}

// A fragment (16x32 MxK, bf16): lane L holds row (L&15); K runs [kb..kb+7] and
// [kb+16..kb+23] with kb = (L>>4)*8.  Two contiguous 16B loads per lane.
__device__ __forceinline__ bf16x16 load_A_frag(const bf16* base, int ld) {
  int lane = threadIdx.x & 31;
  const bf16* p = base + (lane & 15) * ld + ((lane >> 4) << 3);
  bf16x8 lo = *(const bf16x8*)p;
  bf16x8 hi = *(const bf16x8*)(p + 16);
  bf16x16 r;
#pragma unroll
  for (int i = 0; i < 8; ++i) { r[i] = lo[i]; r[i + 8] = hi[i]; }
  return r;
}

// B fragment (32x16 KxN, bf16): lane L holds column (L&15); K runs [kb..kb+7] and
// [kb+16..kb+23] with kb = (L>>4)*8 (same order as A).
__device__ __forceinline__ bf16x16 load_B_frag(const bf16* base, int ld) {
  int lane = threadIdx.x & 31;
  const bf16* p = base + (lane & 15) + ((lane >> 4) << 3) * ld;
  bf16x16 r;
#pragma unroll
  for (int i = 0; i < 8; ++i) { r[i] = p[i * ld]; r[i + 8] = p[(i + 16) * ld]; }
  return r;
}

// ---------------- GroupNorm: (4,128,64,64) -> bf16 (b, token, channel) -------
__global__ void gn_kernel(const float* __restrict__ x, const float* __restrict__ gw,
                          const float* __restrict__ gb, bf16* __restrict__ xn) {
  const int bg = blockIdx.x;            // b*8 + g
  const int b = bg >> 3, g = bg & 7;
  const int tid = threadIdx.x;
  const float* xp = x + (size_t)(b * 128 + g * 16) * 4096;
  float s = 0.f, s2 = 0.f;
  for (int i = tid; i < 16 * 4096; i += 256) {
    float v = xp[i]; s += v; s2 += v * v;
  }
  __shared__ float red[512];
  red[tid] = s; red[256 + tid] = s2;
  __syncthreads();
  for (int off = 128; off > 0; off >>= 1) {
    if (tid < off) { red[tid] += red[tid + off]; red[256 + tid] += red[256 + tid + off]; }
    __syncthreads();
  }
  const float mean = red[0] * (1.f / 65536.f);
  const float var  = red[256] * (1.f / 65536.f) - mean * mean;
  const float rstd = rsqrtf(var + 1e-5f);
  for (int i = tid; i < 16 * 4096; i += 256) {
    int lc = i >> 12;                 // local channel 0..15
    int n  = i & 4095;                // pixel index
    int ch = g * 16 + lc;
    float v = (xp[i] - mean) * rstd * gw[ch] + gb[ch];
    xn[((size_t)(b * 4096 + n)) * 128 + ch] = (bf16)v;
  }
}

// ------------- weight prep: transpose + bf16 convert -------------------------
__global__ void prep_weights(const float* __restrict__ qkvw, const float* __restrict__ projw,
                             bf16* __restrict__ wT, bf16* __restrict__ pT) {
  int idx = blockIdx.x * 256 + threadIdx.x;
  if (idx < 384 * 128) {
    int o = idx / 128, c = idx % 128;
    wT[c * 384 + o] = (bf16)qkvw[idx];
  }
  if (idx < 128 * 128) {
    int o = idx / 128, c = idx % 128;
    pT[c * 128 + o] = (bf16)projw[idx];
  }
}

// ---------------- QKV GEMM: (4096x384) = xn(4096x128) * wT(128x384) ----------
__global__ void qkv_kernel(const bf16* __restrict__ xn, const bf16* __restrict__ wT,
                           const float* __restrict__ qkv_bias,
                           bf16* __restrict__ q, bf16* __restrict__ kT,
                           bf16* __restrict__ v) {
  const int b    = blockIdx.z;
  const int wave = threadIdx.x >> 5;
  const int lane = threadIdx.x & 31;
  const int t0   = (blockIdx.y * 8 + wave) * 16;
  const int o0   = blockIdx.x * 16;
  const bf16* Abase = xn + ((size_t)(b * 4096 + t0)) * 128;
  f32x8 acc = {};
#pragma unroll
  for (int k = 0; k < 4; ++k) {
    bf16x16 a  = load_A_frag(Abase + k * 32, 128);
    bf16x16 bb = load_B_frag(wT + (k * 32) * 384 + o0, 384);
    acc = WMMA_BF16(a, bb, acc);
  }
  const int   n    = lane & 15;
  const int   mo   = (lane >> 4) * 8;
  const int   o    = o0 + n;
  const float bias = qkv_bias[o];
  const float scale = 0.08838834764831845f;   // 128^-0.5, folded into Q
#pragma unroll
  for (int r = 0; r < 8; ++r) {
    int token = t0 + mo + r;
    float val = acc[r] + bias;
    if (o < 128) {
      q[((size_t)(b * 4096 + token)) * 128 + o] = (bf16)(val * scale);
    } else if (o < 256) {
      kT[((size_t)(b * 128 + (o - 128))) * 4096 + token] = (bf16)val;   // transposed
    } else {
      v[((size_t)(b * 4096 + token)) * 128 + (o - 256)] = (bf16)val;
    }
  }
}

// ---------------- Flash attention: O = softmax(Q K^T) V ----------------------
// 128 threads = 4 waves, 64 queries per block; K/V tiles double-buffered in
// LDS, filled with GLOBAL_LOAD_ASYNC_TO_LDS_B128 (ASYNCcnt) so DMA of tile
// i+1 overlaps the 32 WMMAs + softmax of tile i.
__global__ void attn_kernel(const bf16* __restrict__ q, const bf16* __restrict__ kT,
                            const bf16* __restrict__ v, bf16* __restrict__ obuf) {
  const int b    = blockIdx.y;
  const int wave = threadIdx.x >> 5;
  const int lane = threadIdx.x & 31;
  const int tid  = threadIdx.x;
  const int qt   = blockIdx.x * 64 + wave * 16;

  __shared__ __align__(32) bf16 ktile[2][128][64];   // (channel, key)
  __shared__ __align__(32) bf16 vtile[2][64][128];   // (key, channel)
  __shared__ __align__(32) bf16 Plds[4][16][64];     // per-wave P staging

  const bf16* qb = q + ((size_t)(b * 4096 + qt)) * 128;
  bf16x16 aq[4];
#pragma unroll
  for (int k = 0; k < 4; ++k) aq[k] = load_A_frag(qb + k * 32, 128);

  f32x8 zero = {};
  f32x8 of[8];
#pragma unroll
  for (int f = 0; f < 8; ++f) of[f] = zero;
  float mrow[8], lrow[8];
#pragma unroll
  for (int r = 0; r < 8; ++r) { mrow[r] = -INFINITY; lrow[r] = 0.f; }

  const bf16* kTb = kT + (size_t)b * 128 * 4096;
  const bf16* vb  = v  + (size_t)b * 4096 * 128;

  // per-thread staging slices: one K channel-row (128B) + half a V token-row
  const int kc    = tid;               // 0..127 channel
  const int vt    = tid >> 1;          // 0..63 token
  const int vhalf = tid & 1;           // 0/1 -> 64 channels each

  auto stage = [&](int buf, int kt0) {
    const bf16* ks = kTb + (size_t)kc * 4096 + kt0;
    unsigned    kd = (unsigned)(size_t)&ktile[buf][kc][0];
#pragma unroll
    for (int j = 0; j < 8; ++j) async_ld16(kd + j * 16, ks + j * 8);
    const bf16* vs = vb + (size_t)(kt0 + vt) * 128 + vhalf * 64;
    unsigned    vd = (unsigned)(size_t)&vtile[buf][vt][vhalf * 64];
#pragma unroll
    for (int j = 0; j < 8; ++j) async_ld16(vd + j * 16, vs + j * 8);
  };

  stage(0, 0);                          // 16 asyncs outstanding
  for (int it = 0; it < 64; ++it) {
    const int buf = it & 1;
    if (it + 1 < 64) {
      stage(buf ^ 1, (it + 1) * 64);    // prefetch next tile (32 outstanding)
      wait_async_le16();                // drain current tile only (in-order)
    } else {
      wait_async_0();
    }
    __syncthreads();

    // S (16 queries x 64 keys) = Q * K^T, K-reduction over 128 channels
    f32x8 sf[4];
#pragma unroll
    for (int f = 0; f < 4; ++f) {
      sf[f] = zero;
#pragma unroll
      for (int k = 0; k < 4; ++k) {
        bf16x16 bb = load_B_frag(&ktile[buf][k * 32][f * 16], 64);
        sf[f] = WMMA_BF16(aq[k], bb, sf[f]);
      }
    }

    // online softmax; row m = r + 8*(lane>>4) lives across 16 lanes
    {
      const int nlane = lane & 15;
      const int mo    = (lane >> 4) * 8;
#pragma unroll
      for (int r = 0; r < 8; ++r) {
        float mx = fmaxf(fmaxf(sf[0][r], sf[1][r]), fmaxf(sf[2][r], sf[3][r]));
#pragma unroll
        for (int d = 8; d >= 1; d >>= 1) mx = fmaxf(mx, __shfl_xor(mx, d));
        float mnew = fmaxf(mrow[r], mx);
        float corr = __expf(mrow[r] - mnew);
        mrow[r] = mnew;
        lrow[r] *= corr;
#pragma unroll
        for (int f = 0; f < 8; ++f) of[f][r] *= corr;
        float ps = 0.f;
#pragma unroll
        for (int f = 0; f < 4; ++f) {
          float p = __expf(sf[f][r] - mnew);
          Plds[wave][mo + r][f * 16 + nlane] = (bf16)p;
          ps += p;
        }
#pragma unroll
        for (int d = 8; d >= 1; d >>= 1) ps += __shfl_xor(ps, d);
        lrow[r] += ps;
      }
    }
    __syncthreads();

    // O += P (16x64) * V (64x128)
#pragma unroll
    for (int ks = 0; ks < 2; ++ks) {
      bf16x16 ap = load_A_frag(&Plds[wave][0][ks * 32], 64);
#pragma unroll
      for (int f = 0; f < 8; ++f) {
        bf16x16 bb = load_B_frag(&vtile[buf][ks * 32][f * 16], 128);
        of[f] = WMMA_BF16(ap, bb, of[f]);
      }
    }
    __syncthreads();
  }

  // normalize and store O as bf16 (token, channel)
  const int nlane = lane & 15;
  const int mo    = (lane >> 4) * 8;
#pragma unroll
  for (int r = 0; r < 8; ++r) {
    float inv = 1.f / lrow[r];
    int token = qt + mo + r;
#pragma unroll
    for (int f = 0; f < 8; ++f) {
      obuf[((size_t)(b * 4096 + token)) * 128 + f * 16 + nlane] =
          (bf16)(of[f][r] * inv);
    }
  }
}

// ---------------- Proj GEMM + bias + residual, f32 out (b,c,h,w) -------------
__global__ void proj_kernel(const bf16* __restrict__ obuf, const bf16* __restrict__ pT,
                            const float* __restrict__ pbias, const float* __restrict__ x,
                            float* __restrict__ out) {
  const int b    = blockIdx.z;
  const int wave = threadIdx.x >> 5;
  const int lane = threadIdx.x & 31;
  const int t0   = (blockIdx.y * 8 + wave) * 16;
  const int co0  = blockIdx.x * 16;
  const bf16* Abase = obuf + ((size_t)(b * 4096 + t0)) * 128;
  f32x8 acc = {};
#pragma unroll
  for (int k = 0; k < 4; ++k) {
    bf16x16 a  = load_A_frag(Abase + k * 32, 128);
    bf16x16 bb = load_B_frag(pT + (k * 32) * 128 + co0, 128);
    acc = WMMA_BF16(a, bb, acc);
  }
  const int   n    = lane & 15;
  const int   mo   = (lane >> 4) * 8;
  const int   co   = co0 + n;
  const float bias = pbias[co];
  const size_t base = (size_t)b * 128 * 4096 + (size_t)co * 4096;
#pragma unroll
  for (int r = 0; r < 8; ++r) {
    int token = t0 + mo + r;
    out[base + token] = acc[r] + bias + x[base + token];
  }
}

extern "C" void kernel_launch(void* const* d_in, const int* in_sizes, int n_in,
                              void* d_out, int out_size, void* d_ws, size_t ws_size,
                              hipStream_t stream) {
  const float* x     = (const float*)d_in[0];
  const float* gw    = (const float*)d_in[1];
  const float* gb    = (const float*)d_in[2];
  const float* qkvw  = (const float*)d_in[3];
  const float* qkvb  = (const float*)d_in[4];
  const float* projw = (const float*)d_in[5];
  const float* projb = (const float*)d_in[6];
  float* out = (float*)d_out;

  char* ws = (char*)d_ws;
  size_t off = 0;
  auto alloc = [&](size_t bytes) -> char* {
    char* p = ws + off;
    off += (bytes + 255) & ~(size_t)255;
    return p;
  };
  bf16* xn   = (bf16*)alloc(4ull * 4096 * 128 * 2);
  bf16* qbuf = (bf16*)alloc(4ull * 4096 * 128 * 2);
  bf16* kTb  = (bf16*)alloc(4ull * 128 * 4096 * 2);
  bf16* vbuf = (bf16*)alloc(4ull * 4096 * 128 * 2);
  bf16* obuf = (bf16*)alloc(4ull * 4096 * 128 * 2);
  bf16* wT   = (bf16*)alloc(128ull * 384 * 2);
  bf16* pT   = (bf16*)alloc(128ull * 128 * 2);

  hipLaunchKernelGGL(gn_kernel, dim3(32), dim3(256), 0, stream, x, gw, gb, xn);
  hipLaunchKernelGGL(prep_weights, dim3(192), dim3(256), 0, stream, qkvw, projw, wT, pT);
  hipLaunchKernelGGL(qkv_kernel, dim3(24, 32, 4), dim3(256), 0, stream,
                     xn, wT, qkvb, qbuf, kTb, vbuf);
  hipLaunchKernelGGL(attn_kernel, dim3(64, 4), dim3(128), 0, stream,
                     qbuf, kTb, vbuf, obuf);
  hipLaunchKernelGGL(proj_kernel, dim3(8, 32, 4), dim3(256), 0, stream,
                     obuf, pT, projb, x, out);
}
